// MOC_Branch_KwithM_11553462026512
// MI455X (gfx1250) — matrix operand order repaired and checked by reading, not compiled, not run on hardware
//
#include <hip/hip_runtime.h>
#include <stdint.h>

// Problem constants
#define B_    2
#define KF_   7
#define C_    256
#define HH_   36
#define WW_   36
#define HW_   1296
#define N_    9072     // KF_*HW_
#define NP_   9088     // N_ padded to multiple of 32
#define IC_   128
#define HC_   256
#define NT16_ 568      // NP_/16
#define NT32_ 284      // NP_/32
#define RT_   567      // N_/16 valid query tiles

typedef unsigned short u16;
typedef __attribute__((ext_vector_type(16))) __bf16 v16bf;
typedef __attribute__((ext_vector_type(8)))  float  v8f;

// 32-byte fragment: 16 bf16 values in a lane (A or B operand of one WMMA)
union FragU { v16bf v; u16 u[16]; uint4 q[2]; };

__device__ inline u16 f2bf(float f) {
  union { float f; uint32_t u; } cv; cv.f = f;
  uint32_t x = cv.u;
  uint32_t r = (x + 0x7FFFu + ((x >> 16) & 1u)) >> 16;  // RNE
  return (u16)r;
}
__device__ inline float bf2f(u16 h) {
  union { uint32_t u; float f; } cv; cv.u = ((uint32_t)h) << 16;
  return cv.f;
}

// 16-bit A-matrix 16x32 element->K map used here: contiguous k = 16*hl + j (hl = lane/16)
__device__ inline int kmapA(int j, int hl) {
  return hl * 16 + j;
}

__device__ inline v8f wmma_bf16(const FragU& a, const FragU& b, v8f c) {
  return __builtin_amdgcn_wmma_f32_16x16x32_bf16(false, a.v, false, b.v,
                                                 (short)0, c, false, false);
}

// 16 contiguous bf16 -> fragment (2 x b128)
__device__ inline FragU ldfrag(const u16* p) {
  FragU u;
  u.q[0] = *(const uint4*)(p);
  u.q[1] = *(const uint4*)(p + 8);
  return u;
}
// two 8-element halves, 16 elements apart (A-frag from row-major 32-K slice)
__device__ inline FragU ldfrag_split(const u16* p) {
  FragU u;
  u.q[0] = *(const uint4*)(p);
  u.q[1] = *(const uint4*)(p + 16);
  return u;
}

// ---------------------------------------------------------------------------
// Swizzle f32 weights (O x CI [x TAPS]) into A-fragment-major bf16:
// dst[(((oT*KT + kt)*TAPS + tap)*32 + lane)*16 + j] = w[(o*CI + c)*TAPS + tap]
// with o = oT*16 + lane%16, c = kt*32 + kmapA(j, lane/16)
__global__ __launch_bounds__(256)
void swzA_kernel(const float* __restrict__ w, u16* __restrict__ dst,
                 int O, int CI, int TAPS) {
  int idx = blockIdx.x * blockDim.x + threadIdx.x;
  int total = O * CI * TAPS;
  if (idx >= total) return;
  int j    = idx & 15;
  int lane = (idx >> 4) & 31;
  int r    = idx >> 9;
  int tap  = r % TAPS;  r /= TAPS;
  int KT   = CI >> 5;
  int kt   = r % KT;
  int oT   = r / KT;
  int hl = lane >> 4, col = lane & 15;
  int o = oT * 16 + col;
  int c = kt * 32 + kmapA(j, hl);
  dst[idx] = f2bf(w[(o * CI + c) * TAPS + tap]);
}

// ---------------------------------------------------------------------------
// xn in B-fragment-major layout:
// xnS[(((b*NT16 + nT)*8 + kt)*32 + lane)*16 + j] = xn[b][c = kt*32+hl*16+j][n = nT*16+col]
__global__ __launch_bounds__(256)
void prep_xn_kernel(const float* __restrict__ x, u16* __restrict__ xnS) {
  int idx = blockIdx.x * blockDim.x + threadIdx.x;
  int total = B_ * NT16_ * 8 * 512;
  if (idx >= total) return;
  int j    = idx & 15;
  int lane = (idx >> 4) & 31;
  int r    = idx >> 9;
  int kt = r % 8;  r /= 8;
  int nT = r % NT16_;
  int b  = r / NT16_;
  int hl = lane >> 4, col = lane & 15;
  int c = kt * 32 + hl * 16 + j;
  int n = nT * 16 + col;
  float v = 0.f;
  if (n < N_) {
    int k = n / HW_, hw = n % HW_;
    v = x[((b * KF_ + k) * C_ + c) * HW_ + hw];
  }
  xnS[idx] = f2bf(v);
}

// ---------------------------------------------------------------------------
// Embedding GEMM: D(128 x NP) = W(128x256) @ xn + bias, one wave per 16x16 tile.
// mode 0: th row-major (B, NP, 128)
// mode 1: phS B-fragment layout for attention scores
// mode 2: gS  B-fragment layout for attention P@G accumulation
__global__ __launch_bounds__(256)
void embed_gemm_kernel(const u16* __restrict__ xnS, const u16* __restrict__ wA,
                       const float* __restrict__ bias, u16* __restrict__ out,
                       int mode) {
  int gwave = (blockIdx.x * blockDim.x + threadIdx.x) >> 5;
  int lane = threadIdx.x & 31;
  int hl = lane >> 4, col = lane & 15;
  int b = gwave / (8 * NT16_);
  if (b >= B_) return;
  int t = gwave % (8 * NT16_);
  int oT = t / NT16_, nT = t % NT16_;

  v8f acc = {};
  for (int kt = 0; kt < 8; ++kt) {
    FragU a  = ldfrag(wA + ((oT * 8 + kt) * 32 + lane) * 16);
    FragU bm = ldfrag(xnS + (((b * NT16_ + nT) * 8 + kt) * 32 + lane) * 16);
    acc = wmma_bf16(a, bm, acc);
  }
#pragma unroll
  for (int v = 0; v < 8; ++v) {
    int o = oT * 16 + v + 8 * hl;
    int n = nT * 16 + col;
    u16 h = f2bf(acc[v] + bias[o]);
    if (mode == 0) {
      out[((size_t)b * NP_ + n) * IC_ + o] = h;
    } else if (mode == 1) {
      // phS[b][nT][kk=o>>5][lane'=((o>>4)&1)*16+col][j=o&15]
      int idx = (((b * NT16_ + nT) * 4 + (oT >> 1)) * 32 + (oT & 1) * 16 + col) * 16
                + (v + 8 * hl);
      out[idx] = h;
    } else {
      // gS[b][n>>5][t=o>>4][lane'=((n>>4)&1)*16+(o&15)][j=n&15]
      int idx = (((b * NT32_ + (n >> 5)) * 8 + oT) * 32 + ((n >> 4) & 1) * 16
                 + (v + 8 * hl)) * 16 + (n & 15);
      out[idx] = h;
    }
  }
}

// ---------------------------------------------------------------------------
// Fused flash-style attention. One wave per 16-row query tile; 8 independent
// waves per block.  All phi/g operands are 2xb128 fragment loads; only the
// P (scores C-layout -> A-layout) transpose goes through LDS.
__global__ __launch_bounds__(256)
void attention_kernel(const u16* __restrict__ thB, const u16* __restrict__ phS,
                      const u16* __restrict__ gS, u16* __restrict__ ybS) {
  __shared__ __attribute__((aligned(16))) u16 pT[8][16 * 32];
  int tid = threadIdx.x;
  int wave = tid >> 5, lane = tid & 31;
  int hl = lane >> 4, col = lane & 15;
  int gw = blockIdx.x * 8 + wave;
  if (gw >= B_ * RT_) return;
  int b  = gw / RT_;
  int rT = gw % RT_;

  const u16* th_b = thB + (size_t)b * NP_ * IC_;
  const u16* ph_b = phS + (size_t)b * NT16_ * 4 * 512;
  const u16* g_b  = gS  + (size_t)b * NT32_ * 8 * 512;

  // 16x128 query block as 4 A-fragments (vectorized: 16 contiguous at hl*16)
  FragU thA[4];
#pragma unroll
  for (int kk = 0; kk < 4; ++kk)
    thA[kk] = ldfrag(th_b + (rT * 16 + col) * IC_ + kk * 32 + hl * 16);

  v8f zerov = {};
  float m[8], l[8];
  v8f acc[8];
#pragma unroll
  for (int i = 0; i < 8; ++i) { m[i] = -1e30f; l[i] = 0.f; acc[i] = zerov; }

  for (int jt = 0; jt < NT32_; ++jt) {
    if (jt + 1 < NT32_) {
      __builtin_prefetch(ph_b + ((size_t)(jt + 1) * 2 * 4 * 32 + lane) * 16, 0, 1);
      __builtin_prefetch(g_b + ((size_t)(jt + 1) * 8 * 32 + lane) * 16, 0, 1);
    }
    // scores: two 16x16 C-tiles over K=128
    v8f c0 = zerov, c1 = zerov;
#pragma unroll
    for (int kk = 0; kk < 4; ++kk) {
      FragU b0 = ldfrag(ph_b + (((jt * 2 + 0) * 4 + kk) * 32 + lane) * 16);
      FragU b1 = ldfrag(ph_b + (((jt * 2 + 1) * 4 + kk) * 32 + lane) * 16);
      c0 = wmma_bf16(thA[kk], b0, c0);
      c1 = wmma_bf16(thA[kk], b1, c1);
    }
    int n0 = jt * 32;
    if (n0 + col >= N_) {
#pragma unroll
      for (int v = 0; v < 8; ++v) c0[v] = -1e30f;
    }
    if (n0 + 16 + col >= N_) {
#pragma unroll
      for (int v = 0; v < 8; ++v) c1[v] = -1e30f;
    }
    // online softmax (row = v + 8*hl; reduce across the 16 lanes of each half)
#pragma unroll
    for (int v = 0; v < 8; ++v) {
      float sm = fmaxf(c0[v], c1[v]);
#pragma unroll
      for (int off = 1; off < 16; off <<= 1) sm = fmaxf(sm, __shfl_xor(sm, off, 32));
      float mN = fmaxf(m[v], sm);
      float sc = __expf(m[v] - mN);
      float p0 = __expf(c0[v] - mN);
      float p1 = __expf(c1[v] - mN);
      float rs = p0 + p1;
#pragma unroll
      for (int off = 1; off < 16; off <<= 1) rs += __shfl_xor(rs, off, 32);
      m[v] = mN;
      l[v] = l[v] * sc + rs;
#pragma unroll
      for (int t = 0; t < 8; ++t) acc[t][v] *= sc;
      int row = v + 8 * hl;
      pT[wave][row * 32 + col]      = f2bf(p0);
      pT[wave][row * 32 + 16 + col] = f2bf(p1);
    }
    // acc(16x128) += P(16x32) @ G(32x128); DS ops are in-order within a wave
    FragU aP = ldfrag(&pT[wave][col * 32 + hl * 16]);
#pragma unroll
    for (int t = 0; t < 8; ++t) {
      FragU bG = ldfrag(g_b + (((jt * 8) + t) * 32 + lane) * 16);
      acc[t] = wmma_bf16(aP, bG, acc[t]);
    }
  }
  // normalize and store y in B-fragment layout for the wz GEMM
#pragma unroll
  for (int t = 0; t < 8; ++t)
#pragma unroll
    for (int v = 0; v < 8; ++v) {
      int idx = (((b * NT16_ + rT) * 4 + (t >> 1)) * 32 + (t & 1) * 16
                 + (v + 8 * hl)) * 16 + col;
      ybS[idx] = f2bf(acc[t][v] / l[v]);
    }
}

// ---------------------------------------------------------------------------
// wz projection (256x128) + bias + BN + residual -> z in B-fragment layout
__global__ __launch_bounds__(256)
void wz_bn_res_kernel(const u16* __restrict__ ybS, const u16* __restrict__ wA,
                      const float* __restrict__ wz_b, const float* __restrict__ wz_g,
                      const float* __restrict__ wz_be, const float* __restrict__ wz_m,
                      const float* __restrict__ wz_v, const float* __restrict__ x,
                      u16* __restrict__ zS) {
  int gwave = (blockIdx.x * blockDim.x + threadIdx.x) >> 5;
  int lane = threadIdx.x & 31;
  int hl = lane >> 4, col = lane & 15;
  if (gwave >= (B_ * KF_) * 16 * 81) return;
  int bk  = gwave / (16 * 81);
  int rem = gwave % (16 * 81);
  int oT = rem / 81, hwT = rem % 81;
  int b = bk / KF_, k = bk % KF_;
  int nT = k * 81 + hwT;     // 16-column tile index inside (B, N)

  v8f acc = {};
#pragma unroll
  for (int kt = 0; kt < 4; ++kt) {
    FragU a  = ldfrag(wA + ((oT * 4 + kt) * 32 + lane) * 16);
    FragU bm = ldfrag(ybS + (((b * NT16_ + nT) * 4 + kt) * 32 + lane) * 16);
    acc = wmma_bf16(a, bm, acc);
  }
#pragma unroll
  for (int v = 0; v < 8; ++v) {
    int o  = oT * 16 + v + 8 * hl;
    int hw = hwT * 16 + col;
    float val = acc[v] + wz_b[o];
    val = (val - wz_m[o]) * (wz_g[o] * rsqrtf(wz_v[o] + 1e-5f)) + wz_be[o];
    val += x[((size_t)bk * C_ + o) * HW_ + hw];
    // zS[bk][hwT][kt=o>>5][lane'=((o>>4)&1)*16+col][j=o&15]
    int idx = (((bk * 81 + hwT) * 8 + (oT >> 1)) * 32 + (oT & 1) * 16 + col) * 16
              + (v + 8 * hl);
    zS[idx] = f2bf(val);
  }
}

// ---------------------------------------------------------------------------
// shrink 1x1 (256->64) + BN -> s pixel-major (frame, hw, 64)
__global__ __launch_bounds__(256)
void shrink_kernel(const u16* __restrict__ zS, const u16* __restrict__ wA,
                   const float* __restrict__ sh_g, const float* __restrict__ sh_be,
                   const float* __restrict__ sh_m, const float* __restrict__ sh_v,
                   u16* __restrict__ s) {
  int gwave = (blockIdx.x * blockDim.x + threadIdx.x) >> 5;
  int lane = threadIdx.x & 31;
  int hl = lane >> 4, col = lane & 15;
  if (gwave >= (B_ * KF_) * 4 * 81) return;
  int bk  = gwave / (4 * 81);
  int rem = gwave % (4 * 81);
  int oT = rem / 81, hwT = rem % 81;

  v8f acc = {};
#pragma unroll
  for (int kt = 0; kt < 8; ++kt) {
    FragU a  = ldfrag(wA + ((oT * 8 + kt) * 32 + lane) * 16);
    FragU bm = ldfrag(zS + (((bk * 81 + hwT) * 8 + kt) * 32 + lane) * 16);
    acc = wmma_bf16(a, bm, acc);
  }
#pragma unroll
  for (int v = 0; v < 8; ++v) {
    int o  = oT * 16 + v + 8 * hl;
    int hw = hwT * 16 + col;
    float val = (acc[v] - sh_m[o]) * (sh_g[o] * rsqrtf(sh_v[o] + 1e-5f)) + sh_be[o];
    s[((size_t)bk * HW_ + hw) * 64 + o] = f2bf(val);
  }
}

// ---------------------------------------------------------------------------
// 3x3 conv (SAME) + bias + ReLU, implicit GEMM.
// input: s pixel-major; channel cc of logical frame f lives at
//   ((frameBase + cc/64)*HW + pixel)*64 + cc%64   (frameBase = f or f*KF_)
// output: pixel-major (f, hw, 256)
__global__ __launch_bounds__(256)
void conv3x3_relu_kernel(const u16* __restrict__ in, const u16* __restrict__ wA,
                         const float* __restrict__ bias, u16* __restrict__ out,
                         int CI, int F) {
  int gwave = (blockIdx.x * blockDim.x + threadIdx.x) >> 5;
  int lane = threadIdx.x & 31;
  int hl = lane >> 4, col = lane & 15;
  if (gwave >= F * 16 * 81) return;
  int f   = gwave / (16 * 81);
  int rem = gwave % (16 * 81);
  int oT = rem / 81, hwT = rem % 81;
  int hw = hwT * 16 + col;
  int py = hw / WW_, px = hw % WW_;
  int KT = CI >> 5;
  int frameBase = (CI == 64) ? f : f * KF_;

  v8f acc = {};
  for (int dy = -1; dy <= 1; ++dy) {
    for (int dx = -1; dx <= 1; ++dx) {
      int sy = py + dy, sx = px + dx;
      bool ok = (sy >= 0) && (sy < HH_) && (sx >= 0) && (sx < WW_);
      int syc = sy < 0 ? 0 : (sy > HH_ - 1 ? HH_ - 1 : sy);
      int sxc = sx < 0 ? 0 : (sx > WW_ - 1 ? WW_ - 1 : sx);
      int sidx = syc * WW_ + sxc;
      int tap = (dy + 1) * 3 + (dx + 1);
      for (int kt = 0; kt < KT; ++kt) {
        FragU a = ldfrag(wA + (((oT * KT + kt) * 9 + tap) * 32 + lane) * 16);
        int cc = kt * 32 + hl * 16;   // 16 consecutive channels, never crosses 64
        const u16* bp = in + ((size_t)(frameBase + (cc >> 6)) * HW_ + sidx) * 64
                        + (cc & 63);
        FragU bm = ldfrag(bp);
        if (!ok) { bm.q[0] = make_uint4(0, 0, 0, 0); bm.q[1] = make_uint4(0, 0, 0, 0); }
        acc = wmma_bf16(a, bm, acc);
      }
    }
  }
#pragma unroll
  for (int v = 0; v < 8; ++v) {
    int o = oT * 16 + v + 8 * hl;
    float val = fmaxf(acc[v] + bias[o], 0.f);
    out[((size_t)f * HW_ + hw) * HC_ + o] = f2bf(val);
  }
}

// ---------------------------------------------------------------------------
// tiny 1x1 heads (2/24/14 out-ch, K=256); conv outputs are pixel-major so the
// K loop reads 256 contiguous bf16.
__global__ __launch_bounds__(256)
void head1x1_kernel(const u16* __restrict__ in, const float* __restrict__ w,
                    const float* __restrict__ bias, float* __restrict__ out,
                    int F, int OC, int act) {
  int idx = blockIdx.x * blockDim.x + threadIdx.x;
  int total = F * OC * HW_;
  if (idx >= total) return;
  int hw = idx % HW_;
  int t  = idx / HW_;
  int oc = t % OC;
  int f  = t / OC;
  float acc = bias[oc];
  const u16* ip = in + ((size_t)f * HW_ + hw) * HC_;
  const float* wp = w + (size_t)oc * HC_;
  for (int c = 0; c < HC_; ++c) acc += bf2f(ip[c]) * wp[c];
  if (act) acc = 1.f / (1.f + __expf(-acc));
  out[idx] = acc;
}

// ---------------------------------------------------------------------------
extern "C" void kernel_launch(void* const* d_in, const int* in_sizes, int n_in,
                              void* d_out, int out_size, void* d_ws, size_t ws_size,
                              hipStream_t stream) {
  (void)in_sizes; (void)n_in; (void)out_size; (void)ws_size;

  const float* x     = (const float*)d_in[0];
  const float* g_w   = (const float*)d_in[1];
  const float* g_b   = (const float*)d_in[2];
  const float* th_w  = (const float*)d_in[3];
  const float* th_b  = (const float*)d_in[4];
  const float* ph_w  = (const float*)d_in[5];
  const float* ph_b  = (const float*)d_in[6];
  const float* wz_w  = (const float*)d_in[7];
  const float* wz_b  = (const float*)d_in[8];
  const float* wz_g  = (const float*)d_in[9];
  const float* wz_be = (const float*)d_in[10];
  const float* wz_m  = (const float*)d_in[11];
  const float* wz_v  = (const float*)d_in[12];
  const float* sh_w  = (const float*)d_in[13];
  const float* sh_g  = (const float*)d_in[14];
  const float* sh_be = (const float*)d_in[15];
  const float* sh_m  = (const float*)d_in[16];
  const float* sh_v  = (const float*)d_in[17];
  const float* wh1_w = (const float*)d_in[18];
  const float* wh1_b = (const float*)d_in[19];
  const float* wh2_w = (const float*)d_in[20];
  const float* wh2_b = (const float*)d_in[21];
  const float* hm1_w = (const float*)d_in[22];
  const float* hm1_b = (const float*)d_in[23];
  const float* hmc_w = (const float*)d_in[24];
  const float* hmc_b = (const float*)d_in[25];
  const float* mv1_w = (const float*)d_in[26];
  const float* mv1_b = (const float*)d_in[27];
  const float* mv2_w = (const float*)d_in[28];
  const float* mv2_b = (const float*)d_in[29];

  char* ws = (char*)d_ws;
  u16* xnS = (u16*)(ws + 0);          // B-fragment xn        9,306,112 B
  u16* gS  = (u16*)(ws + 9306112);    // g fragments          4,653,056 B
  u16* th  = (u16*)(ws + 13959168);   // th row-major         4,653,056 B
  u16* phS = (u16*)(ws + 18612224);   // phi fragments        4,653,056 B
  u16* ybS = (u16*)(ws + 23265280);   // y fragments          4,653,056 B
  u16* zS  = (u16*)(ws + 27918336);   // z fragments          9,289,728 B
  u16* s   = (u16*)(ws + 37208064);   // s pixel-major        2,322,432 B
  u16* whc = (u16*)(ws + 39530496);   // wh conv out          9,289,728 B
  u16* hmc = (u16*)(ws + 48820224);   // hm conv out          1,327,104 B
  u16* mvc = (u16*)(ws + 50147328);   // mv conv out          1,327,104 B
  char* wbase = ws + 51474432;
  u16* wgA   = (u16*)(wbase + 0);
  u16* wthA  = (u16*)(wbase + 65536);
  u16* wphA  = (u16*)(wbase + 131072);
  u16* wwzA  = (u16*)(wbase + 196608);
  u16* wshA  = (u16*)(wbase + 262144);
  u16* wwh1A = (u16*)(wbase + 294912);
  u16* whm1A = (u16*)(wbase + 589824);
  u16* wmv1A = (u16*)(wbase + 2654208);

  // weight swizzles into A-fragment-major bf16
  swzA_kernel<<<128, 256, 0, stream>>>(g_w,  wgA,  128, 256, 1);
  swzA_kernel<<<128, 256, 0, stream>>>(th_w, wthA, 128, 256, 1);
  swzA_kernel<<<128, 256, 0, stream>>>(ph_w, wphA, 128, 256, 1);
  swzA_kernel<<<128, 256, 0, stream>>>(wz_w, wwzA, 256, 128, 1);
  swzA_kernel<<<64, 256, 0, stream>>>(sh_w, wshA, 64, 256, 1);
  swzA_kernel<<<576, 256, 0, stream>>>(wh1_w, wwh1A, 256, 64, 9);
  swzA_kernel<<<4032, 256, 0, stream>>>(hm1_w, whm1A, 256, 448, 9);
  swzA_kernel<<<4032, 256, 0, stream>>>(mv1_w, wmv1A, 256, 448, 9);

  // xn in B-fragment layout (zero pad beyond N)
  prep_xn_kernel<<<(B_ * NT16_ * 8 * 512 + 255) / 256, 256, 0, stream>>>(x, xnS);

  // embedding GEMMs (2 * 8 * 568 = 9088 waves = 1136 blocks)
  embed_gemm_kernel<<<1136, 256, 0, stream>>>(xnS, wthA, th_b, th,  0);
  embed_gemm_kernel<<<1136, 256, 0, stream>>>(xnS, wphA, ph_b, phS, 1);
  embed_gemm_kernel<<<1136, 256, 0, stream>>>(xnS, wgA,  g_b,  gS,  2);

  // fused attention: 1134 query-tile waves -> 142 blocks of 8 waves
  attention_kernel<<<142, 256, 0, stream>>>(th, phS, gS, ybS);

  // wz + BN + residual: 14*16*81 = 18144 waves = 2268 blocks
  wz_bn_res_kernel<<<2268, 256, 0, stream>>>(ybS, wwzA, wz_b, wz_g, wz_be, wz_m,
                                             wz_v, x, zS);

  // shrink: 14*4*81 = 4536 waves = 567 blocks
  shrink_kernel<<<567, 256, 0, stream>>>(zS, wshA, sh_g, sh_be, sh_m, sh_v, s);

  // 3x3 conv heads (implicit GEMM, fully vectorized fragments)
  conv3x3_relu_kernel<<<2268, 256, 0, stream>>>(s, wwh1A, wh1_b, whc, 64, 14);
  conv3x3_relu_kernel<<<324, 256, 0, stream>>>(s, whm1A, hm1_b, hmc, 448, 2);
  conv3x3_relu_kernel<<<324, 256, 0, stream>>>(s, wmv1A, mv1_b, mvc, 448, 2);

  // 1x1 heads -> d_out, concatenated (hm, mv, wh)
  float* out = (float*)d_out;
  head1x1_kernel<<<(62208 + 255) / 256, 256, 0, stream>>>(hmc, hmc_w, hmc_b, out, 2, 24, 1);
  head1x1_kernel<<<(36288 + 255) / 256, 256, 0, stream>>>(mvc, mv2_w, mv2_b, out + 62208, 2, 14, 0);
  head1x1_kernel<<<(36288 + 255) / 256, 256, 0, stream>>>(whc, wh2_w, wh2_b, out + 98496, 14, 2, 0);
}
